// LIFEmbedding_37074157699160
// MI455X (gfx1250) — matrix operand order repaired and checked by reading, not compiled, not run on hardware
//
#include <hip/hip_runtime.h>

typedef _Float16 v16h __attribute__((ext_vector_type(16)));
typedef _Float16 v8h  __attribute__((ext_vector_type(8)));
typedef float    v8f  __attribute__((ext_vector_type(8)));

namespace {
constexpr float DECAY = 0.2f;
constexpr float VTH   = 1.0f;
constexpr int T_STEPS = 8;
constexpr int CIN  = 2;
constexpr int COUT = 64;
constexpr int IH = 128;
constexpr int IW = 128;
constexpr int NB = 8;

// Tile geometry: 2 rows x 16 cols of output pixels per workgroup.
constexpr int TW = 16;
constexpr int THT = 2;
constexpr int H1R = THT + 2;   // 4  (conv2 halo)
constexpr int H1C = TW + 2;    // 18
constexpr int EVR = THT + 4;   // 6  (conv1+conv2 halo)
constexpr int EVC = TW + 4;    // 20
}

__global__ __launch_bounds__(256)
void lif_conv_fused(const float* __restrict__ events,
                    const float* __restrict__ W1,
                    const float* __restrict__ b1,
                    const float* __restrict__ W2,
                    const float* __restrict__ b2,
                    float* __restrict__ out)
{
    __shared__ __align__(16) _Float16 h1[H1R * H1C * COUT];  // 9216 B, [pixel][ch] f16
    __shared__ float evt[EVR * EVC * CIN];                    // 960 B
    __shared__ float w1s[COUT * CIN * 9];                     // 4608 B
    __shared__ float b1s[COUT];

    const int tid   = threadIdx.x;
    const int lane  = tid & 31;        // wave32
    const int wave  = tid >> 5;        // 0..7
    const int hi    = lane >> 4;       // lane half (0/1)
    const int lm    = lane & 15;
    const int strip = wave >> 2;       // row within tile (0/1)
    const int ntile = wave & 3;        // cout block (0..3)

    const int c0 = blockIdx.x * TW;
    const int r0 = blockIdx.y * THT;
    const int b  = blockIdx.z;

    // ---- conv1 weights -> LDS (once) ----
    for (int i = tid; i < COUT * CIN * 9; i += 256) w1s[i] = W1[i];
    if (tid < COUT) b1s[tid] = b1[tid];

    // ---- conv2 weights -> f16 registers (B matrices, 9 taps x 2 K-chunks) ----
    // B layout (16-bit, 32x16): lane holds N = lane%16; element e -> K = (e%8) + (e/8)*16 + 8*hi.
    const int cout = ntile * 16 + lm;
    const int hi8  = hi * 8;
    v16h breg[9][2];
#pragma unroll
    for (int tap = 0; tap < 9; ++tap) {
        const int ky = tap / 3, kx = tap % 3;
#pragma unroll
        for (int kc = 0; kc < 2; ++kc) {
#pragma unroll
            for (int e = 0; e < 16; ++e) {
                const int cin = kc * 32 + hi8 + (e & 7) + ((e >> 3) << 4);
                breg[tap][kc][e] =
                    (_Float16)W2[((cout * COUT + cin) * 3 + ky) * 3 + kx];
            }
        }
    }
    const float bias2 = b2[cout];

    // LIF state lives in the C-matrix register layout across time steps.
    v8f vmem, osum;
#pragma unroll
    for (int i = 0; i < 8; ++i) { vmem[i] = 0.f; osum[i] = 0.f; }

    __syncthreads();

    for (int s = 0; s < T_STEPS; ++s) {
        const int tsrc = T_STEPS - 1 - s;  // time reversal (flip)

        // ---- P1: stage zero-padded events tile ----
        if (tid < EVR * EVC * CIN) {
            const int r  = tid / (EVC * CIN);
            const int rm = tid % (EVC * CIN);
            const int c  = rm >> 1;
            const int ci = rm & 1;
            const int gy = r0 - 2 + r;
            const int gx = c0 - 2 + c;
            float v = 0.f;
            const size_t base = (((size_t)b * T_STEPS + tsrc) * CIN + ci) * (size_t)(IH * IW);
            if (gy >= 0 && gy < IH && gx >= 0 && gx < IW)
                v = events[base + gy * IW + gx];
            evt[(r * EVC + c) * CIN + ci] = v;
            // prefetch next timestep's tile into cache
            if (s + 1 < T_STEPS && gy >= 0 && gy < IH && gx >= 0 && gx < IW) {
                const size_t nbase = (((size_t)b * T_STEPS + (tsrc - 1)) * CIN + ci) * (size_t)(IH * IW);
                __builtin_prefetch(&events[nbase + gy * IW + gx], 0, 0);
            }
        }
        __syncthreads();

        // ---- P2: conv1 + ReLU -> LDS h1 (f16), zero outside the image ----
        {
            const int ch = tid & 63;
            const int pg = tid >> 6;
            for (int i = 0; i < (H1R * H1C) / 4; ++i) {   // 18 pixels per thread
                const int p = pg + 4 * i;                  // 0..71
                const int r = p / H1C, c = p % H1C;
                float acc = b1s[ch];
#pragma unroll
                for (int ci = 0; ci < CIN; ++ci)
#pragma unroll
                    for (int ky = 0; ky < 3; ++ky)
#pragma unroll
                        for (int kx = 0; kx < 3; ++kx)
                            acc += evt[((r + ky) * EVC + (c + kx)) * CIN + ci] *
                                   w1s[((ch * CIN + ci) * 3 + ky) * 3 + kx];
                const int gy = r0 - 1 + r, gx = c0 - 1 + c;
                const bool inb = (gy >= 0) & (gy < IH) & (gx >= 0) & (gx < IW);
                const float v = inb ? fmaxf(acc, 0.f) : 0.f;
                h1[p * COUT + ch] = (_Float16)v;
            }
        }
        __syncthreads();

        // ---- P3: conv2 via v_wmma_f32_16x16x32_f16, software-pipelined ----
        v8f h2;
#pragma unroll
        for (int i = 0; i < 8; ++i) h2[i] = bias2;

        // A layout (16-bit, 16x32): lane M = lane%16; element e ->
        // K = (e%8) + (e/8)*16 + 8*hi  ==> two contiguous 8-channel runs.
        union AV { v16h v; v8h p[2]; };
        AV abuf[2];

        // step = tap*2 + kc, 18 steps total
        auto aptr = [&](int step) -> const _Float16* {
            const int tap = step >> 1, kc = step & 1;
            const int dy = tap / 3 - 1, dx = tap % 3 - 1;
            return &h1[((1 + strip + dy) * H1C + (1 + lm + dx)) * COUT + hi8] + kc * 32;
        };

        {   // prologue: stage step 0
            const _Float16* p0 = aptr(0);
            abuf[0].p[0] = *(const v8h*)(p0);
            abuf[0].p[1] = *(const v8h*)(p0 + 16);
        }
#pragma unroll
        for (int step = 0; step < 18; ++step) {
            if (step + 1 < 18) {   // issue next step's ds_load_b128 pair first
                const _Float16* pn = aptr(step + 1);
                abuf[(step + 1) & 1].p[0] = *(const v8h*)(pn);
                abuf[(step + 1) & 1].p[1] = *(const v8h*)(pn + 16);
            }
            h2 = __builtin_amdgcn_wmma_f32_16x16x32_f16(
                    false, abuf[step & 1].v, false, breg[step >> 1][step & 1],
                    (short)0, h2, false, false);
        }

        // ---- LIF update (elementwise on C tile) ----
#pragma unroll
        for (int i = 0; i < 8; ++i) {
            const float vnr = DECAY * vmem[i] + h2[i];
            osum[i] += vnr;
            vmem[i] = (vnr > VTH) ? 0.f : vnr;
        }
        __syncthreads();
    }

    // ---- store Sum_t v_noreset ----
    // C/D layout: element i -> M = i + 8*hi (pixel/col), N = lane%16 (cout).
    const int orow = r0 + strip;
#pragma unroll
    for (int i = 0; i < 8; ++i) {
        const int col = c0 + i + hi8;
        out[(((size_t)b * COUT + cout) * IH + orow) * IW + col] = osum[i];
    }
}

extern "C" void kernel_launch(void* const* d_in, const int* in_sizes, int n_in,
                              void* d_out, int out_size, void* d_ws, size_t ws_size,
                              hipStream_t stream) {
    const float* events = (const float*)d_in[0];
    const float* W1     = (const float*)d_in[1];
    const float* b1     = (const float*)d_in[2];
    const float* W2     = (const float*)d_in[3];
    const float* b2     = (const float*)d_in[4];
    float* out = (float*)d_out;

    dim3 grid(IW / TW, IH / THT, NB);  // (8, 64, 8)
    dim3 block(256);                   // 8 wave32 waves: 2 strips x 4 cout tiles
    hipLaunchKernelGGL(lif_conv_fused, grid, block, 0, stream,
                       events, W1, b1, W2, b2, out);
}
